// LatentAttentionBlock_72670846648634
// MI455X (gfx1250) — matrix operand order repaired and checked by reading, not compiled, not run on hardware
//
#include <hip/hip_runtime.h>
#include <hip/hip_bf16.h>
#include <math.h>

typedef __bf16 bf16_t;
typedef __attribute__((ext_vector_type(16))) bf16_t v16bf;
typedef __attribute__((ext_vector_type(8)))  bf16_t v8bf;
typedef __attribute__((ext_vector_type(8)))  float  v8f;

// async-to-LDS builtin operand types (per hipcc diagnostic):
//   (int4 addrspace(1)*, int4 addrspace(3)*, imm int, imm int)
typedef int v4i __attribute__((vector_size(16)));
typedef __attribute__((address_space(1))) v4i av4i_g;
typedef __attribute__((address_space(3))) v4i av4i_l;

#if defined(__has_builtin)
#if __has_builtin(__builtin_amdgcn_global_load_async_to_lds_b128) && \
    __has_builtin(__builtin_amdgcn_s_wait_asynccnt)
#define USE_ASYNC_LDS 1
#endif
#endif
#ifndef USE_ASYNC_LDS
#define USE_ASYNC_LDS 0
#endif

#define DDIM   2048
#define QKDIM  256
#define EXPDIM 4096
#define ETOT   8704      // 2*QK + 2*EXP
#define SEQ    2048
#define BATCH  4
#define MROWS  8192      // B*S
#define TS     40        // padded LDS tile row stride (elements): 80B = 20 banks

// ---------------------------------------------------------------- helpers ---
__global__ void cvt_f32_to_bf16(const float* __restrict__ src,
                                bf16_t* __restrict__ dst, long n) {
  long i = (long)blockIdx.x * blockDim.x + threadIdx.x;
  long stride = (long)gridDim.x * blockDim.x;
  for (; i < n; i += stride) dst[i] = (bf16_t)src[i];
}

__global__ __launch_bounds__(256)
void layernorm_bf16(const float* __restrict__ x, const float* __restrict__ g,
                    const float* __restrict__ b, bf16_t* __restrict__ h) {
  __shared__ float red[256];
  const int row = blockIdx.x;
  const float* xr = x + (size_t)row * DDIM;
  float s = 0.f, sq = 0.f;
  for (int c = threadIdx.x; c < DDIM; c += 256) {
    float v = xr[c]; s += v; sq += v * v;
  }
  red[threadIdx.x] = s; __syncthreads();
  for (int o = 128; o > 0; o >>= 1) {
    if (threadIdx.x < o) red[threadIdx.x] += red[threadIdx.x + o];
    __syncthreads();
  }
  const float mu = red[0] * (1.0f / DDIM);
  __syncthreads();
  red[threadIdx.x] = sq; __syncthreads();
  for (int o = 128; o > 0; o >>= 1) {
    if (threadIdx.x < o) red[threadIdx.x] += red[threadIdx.x + o];
    __syncthreads();
  }
  const float var = red[0] * (1.0f / DDIM) - mu * mu;
  const float rs = rsqrtf(var + 1e-5f);
  bf16_t* hr = h + (size_t)row * DDIM;
  for (int c = threadIdx.x; c < DDIM; c += 256)
    hr[c] = (bf16_t)((xr[c] - mu) * rs * g[c] + b[c]);
}

// fused row (8704 f32) -> q / k / geglu_local (concat[0:2048]) / value
__global__ __launch_bounds__(256)
void geglu_split(const float* __restrict__ fused,
                 bf16_t* __restrict__ qb, bf16_t* __restrict__ kb,
                 bf16_t* __restrict__ concat, bf16_t* __restrict__ value) {
  const int row = blockIdx.x;
  const float* f = fused + (size_t)row * ETOT;
  for (int c = threadIdx.x; c < 2 * QKDIM; c += 256) {
    if (c < QKDIM) qb[(size_t)row * QKDIM + c] = (bf16_t)f[c];
    else           kb[(size_t)row * QKDIM + (c - QKDIM)] = (bf16_t)f[c];
  }
  for (int e = threadIdx.x; e < EXPDIM; e += 256) {
    float lin = f[2 * QKDIM + e];
    float pg  = f[2 * QKDIM + EXPDIM + e];
    float gelu = 0.5f * pg * (1.0f + erff(pg * 0.70710678118654752f));
    float gg = lin * gelu;
    if (e < EXPDIM - DDIM) concat[(size_t)row * EXPDIM + e] = (bf16_t)gg;
    else                   value[(size_t)row * DDIM + (e - (EXPDIM - DDIM))] = (bf16_t)gg;
  }
}

// rowwise softmax: scores f32 (len SEQ) -> probs bf16
__global__ __launch_bounds__(256)
void softmax_rows(const float* __restrict__ sc, bf16_t* __restrict__ p) {
  __shared__ float red[256];
  const int row = blockIdx.x;
  const float* sr = sc + (size_t)row * SEQ;
  float mx = -3.0e38f;
  for (int c = threadIdx.x; c < SEQ; c += 256) mx = fmaxf(mx, sr[c]);
  red[threadIdx.x] = mx; __syncthreads();
  for (int o = 128; o > 0; o >>= 1) {
    if (threadIdx.x < o) red[threadIdx.x] = fmaxf(red[threadIdx.x], red[threadIdx.x + o]);
    __syncthreads();
  }
  const float rowmax = red[0];
  __syncthreads();
  float sum = 0.f;
  for (int c = threadIdx.x; c < SEQ; c += 256) sum += expf(sr[c] - rowmax);
  red[threadIdx.x] = sum; __syncthreads();
  for (int o = 128; o > 0; o >>= 1) {
    if (threadIdx.x < o) red[threadIdx.x] += red[threadIdx.x + o];
    __syncthreads();
  }
  const float inv = 1.0f / red[0];
  bf16_t* pr = p + (size_t)row * SEQ;
  for (int c = threadIdx.x; c < SEQ; c += 256)
    pr[c] = (bf16_t)(expf(sr[c] - rowmax) * inv);
}

// ---------------------------------------------------------- WMMA GEMM ------
// MODE 0: C_f32 = A * B^T                      (fused = h @ expand^T)
// MODE 1: C_f32 = scale*A*B^T + causal ALiBi   (scores, batched)
// MODE 2: C_bf16 = A * B     (B row-major KxN, staged transposed; P @ V)
// MODE 3: C_f32 = A * B^T + residual           (project + residual)
template <int MODE>
__global__ __launch_bounds__(256)
void gemm_wmma(const bf16_t* __restrict__ A, const bf16_t* __restrict__ B,
               float* __restrict__ Cf, bf16_t* __restrict__ Cb,
               int M, int N, int K, int lda, int ldb, int ldc,
               long aBatch, long bBatch, long cBatch,
               float scale, const float* __restrict__ posm,
               const float* __restrict__ resid, int cColOff) {
  __shared__ bf16_t As[2][128 * TS];
  __shared__ bf16_t Bs[2][128 * TS];

  const int nBase = blockIdx.x * 128;
  const int mBase = blockIdx.y * 128;
  const int z = blockIdx.z;
  const bf16_t* Ab = A + (size_t)z * aBatch;
  const bf16_t* Bb = B + (size_t)z * bBatch;

  const int tid = threadIdx.x;
  const int lane = tid & 31, wave = tid >> 5;
  const int half = lane >> 4, l16 = lane & 15;
  const int wm = (wave & 3) * 32;   // 4 waves along M, 32 rows each
  const int wn = (wave >> 2) * 64;  // 2 waves along N, 64 cols each

  if (MODE == 1) {
    // block fully above the diagonal -> all masked, skip compute entirely
    if (nBase > mBase + 127) {
      float* Cz = Cf + (size_t)z * cBatch;
      for (int idx = tid; idx < 128 * 128; idx += 256) {
        int m = mBase + (idx >> 7), n = nBase + (idx & 127);
        Cz[(size_t)m * ldc + n] = -1e30f;
      }
      return;
    }
  }

  int Keff = K;
  if (MODE == 2) {  // P[q,k] == 0 for k > q : clamp K loop to the query block
    int lim = mBase + 128;
    Keff = lim < K ? lim : K;
  }

  v8f acc[2][4] = {};

  // ---- fragment build + 8 WMMA per wave per k-step --------------------
  auto computeTile = [&](int buf) {
    v16bf afr[2];
#pragma unroll
    for (int tm = 0; tm < 2; ++tm) {
      const bf16_t* p = &As[buf][(wm + tm * 16 + l16) * TS + half * 8];
      v8bf lo = *(const v8bf*)p;
      v8bf hi = *(const v8bf*)(p + 16);
#pragma unroll
      for (int i = 0; i < 8; ++i) { afr[tm][i] = lo[i]; afr[tm][8 + i] = hi[i]; }
    }
    v16bf bfr[4];
#pragma unroll
    for (int tn = 0; tn < 4; ++tn) {
      const bf16_t* p = &Bs[buf][(wn + tn * 16 + l16) * TS + half * 8];
      v8bf lo = *(const v8bf*)p;
      v8bf hi = *(const v8bf*)(p + 16);
#pragma unroll
      for (int i = 0; i < 8; ++i) { bfr[tn][i] = lo[i]; bfr[tn][8 + i] = hi[i]; }
    }
#pragma unroll
    for (int tm = 0; tm < 2; ++tm)
#pragma unroll
      for (int tn = 0; tn < 4; ++tn)
        acc[tm][tn] = __builtin_amdgcn_wmma_f32_16x16x32_bf16(
            false, afr[tm], false, bfr[tn], (short)0, acc[tm][tn], false, false);
  };

#if USE_ASYNC_LDS
  // ---- CDNA5 path: global_load_async_to_lds_b128, double-buffered -----
  auto stage = [&](int buf, int k0) {
#pragma unroll
    for (int it = 0; it < 2; ++it) {
      int id = tid + it * 256;           // 0..511
      int row = id >> 2, c8 = (id & 3) * 8;
      __builtin_amdgcn_global_load_async_to_lds_b128(
          (av4i_g*)(Ab + (size_t)(mBase + row) * lda + k0 + c8),
          (av4i_l*)&As[buf][row * TS + c8], 0, 0);
    }
    if (MODE != 2) {
#pragma unroll
      for (int it = 0; it < 2; ++it) {
        int id = tid + it * 256;
        int row = id >> 2, c8 = (id & 3) * 8;
        __builtin_amdgcn_global_load_async_to_lds_b128(
            (av4i_g*)(Bb + (size_t)(nBase + row) * ldb + k0 + c8),
            (av4i_l*)&Bs[buf][row * TS + c8], 0, 0);
      }
    } else {
      // V is row-major KxN : transpose while staging (synchronous)
      for (int e = tid; e < 128 * 32; e += 256) {
        int n = e & 127, kk = e >> 7;
        Bs[buf][n * TS + kk] = Bb[(size_t)(k0 + kk) * ldb + nBase + n];
      }
    }
  };

  stage(0, 0);
  int cur = 0;
  for (int k0 = 0; k0 < Keff; k0 += 32) {
    const bool more = (k0 + 32 < Keff);
    if (more) {
      stage(cur ^ 1, k0 + 32);                 // prev barrier => buf^1 is free
      __builtin_amdgcn_s_wait_asynccnt(MODE == 2 ? 2 : 4);  // cur tile landed
    } else {
      __builtin_amdgcn_s_wait_asynccnt(0);
    }
    __syncthreads();       // all waves' cur tile visible
    computeTile(cur);
    __syncthreads();       // everyone done reading cur before it is restaged
    cur ^= 1;
  }
#else
  // ---- fallback: batched register staging (loads issued back-to-back) --
  const int row0 = tid >> 2,          c80 = (tid & 3) * 8;
  const int row1 = (tid + 256) >> 2,  c81 = (tid & 3) * 8;
  for (int k0 = 0; k0 < Keff; k0 += 32) {
    v8bf ra0, ra1, rb0 = {}, rb1 = {};
    bf16_t rbt[16];
    ra0 = *(const v8bf*)(Ab + (size_t)(mBase + row0) * lda + k0 + c80);
    ra1 = *(const v8bf*)(Ab + (size_t)(mBase + row1) * lda + k0 + c81);
    if (MODE == 2) {
#pragma unroll
      for (int j = 0; j < 16; ++j) {
        int e = tid + j * 256;
        int n = e & 127, kk = e >> 7;
        rbt[j] = Bb[(size_t)(k0 + kk) * ldb + nBase + n];
      }
    } else {
      rb0 = *(const v8bf*)(Bb + (size_t)(nBase + row0) * ldb + k0 + c80);
      rb1 = *(const v8bf*)(Bb + (size_t)(nBase + row1) * ldb + k0 + c81);
    }
    if (k0 + 32 < Keff) {
      __builtin_prefetch(Ab + (size_t)(mBase + (tid >> 1)) * lda + k0 + 32, 0, 1);
      if (MODE != 2)
        __builtin_prefetch(Bb + (size_t)(nBase + (tid >> 1)) * ldb + k0 + 32, 0, 1);
    }
    __syncthreads();       // previous compute done before overwrite
    *(v8bf*)&As[0][row0 * TS + c80] = ra0;
    *(v8bf*)&As[0][row1 * TS + c81] = ra1;
    if (MODE == 2) {
#pragma unroll
      for (int j = 0; j < 16; ++j) {
        int e = tid + j * 256;
        int n = e & 127, kk = e >> 7;
        Bs[0][n * TS + kk] = rbt[j];
      }
    } else {
      *(v8bf*)&Bs[0][row0 * TS + c80] = rb0;
      *(v8bf*)&Bs[0][row1 * TS + c81] = rb1;
    }
    __syncthreads();
    computeTile(0);
  }
#endif

  // ---- epilogue ----
  float sp = 0.f;
  if (MODE == 1) sp = log1pf(expf(posm[0]));
#pragma unroll
  for (int tm = 0; tm < 2; ++tm)
#pragma unroll
    for (int tn = 0; tn < 4; ++tn)
#pragma unroll
      for (int r = 0; r < 8; ++r) {
        int m = mBase + wm + tm * 16 + half * 8 + r;
        int n = nBase + wn + tn * 16 + l16;
        float v = acc[tm][tn][r];
        if (MODE == 0) {
          Cf[(size_t)m * ldc + n] = v;
        } else if (MODE == 1) {
          float val = (n <= m) ? (v * scale + sp * (float)(n - m)) : -1e30f;
          (Cf + (size_t)z * cBatch)[(size_t)m * ldc + n] = val;
        } else if (MODE == 2) {
          (Cb + (size_t)z * cBatch)[(size_t)m * ldc + cColOff + n] = (bf16_t)v;
        } else {
          Cf[(size_t)m * ldc + n] = v + resid[(size_t)m * ldc + n];
        }
      }
}

// ------------------------------------------------------------- launcher ----
extern "C" void kernel_launch(void* const* d_in, const int* in_sizes, int n_in,
                              void* d_out, int out_size, void* d_ws, size_t ws_size,
                              hipStream_t stream) {
  (void)in_sizes; (void)n_in; (void)out_size; (void)ws_size;
  const float* x       = (const float*)d_in[0];
  const float* gamma   = (const float*)d_in[1];
  const float* beta    = (const float*)d_in[2];
  const float* expand  = (const float*)d_in[3];
  const float* project = (const float*)d_in[4];
  const float* posm    = (const float*)d_in[5];
  float* out = (float*)d_out;

  char* ws = (char*)d_ws;
  size_t off = 0;
  auto alloc = [&](size_t bytes) -> char* {
    char* p = ws + off;
    off += (bytes + 255) & ~(size_t)255;
    return p;
  };
  bf16_t* wExpand  = (bf16_t*)alloc((size_t)ETOT * DDIM * 2);
  bf16_t* wProject = (bf16_t*)alloc((size_t)DDIM * EXPDIM * 2);
  bf16_t* wH       = (bf16_t*)alloc((size_t)MROWS * DDIM * 2);
  char*   wFusedRaw = alloc((size_t)MROWS * ETOT * 4);     // 285 MB
  float*  wFused   = (float*)wFusedRaw;
  bf16_t* wQ       = (bf16_t*)alloc((size_t)MROWS * QKDIM * 2);
  bf16_t* wK       = (bf16_t*)alloc((size_t)MROWS * QKDIM * 2);
  bf16_t* wConcat  = (bf16_t*)alloc((size_t)MROWS * EXPDIM * 2);
  bf16_t* wValue   = (bf16_t*)alloc((size_t)MROWS * DDIM * 2);
  // fused buffer is dead after geglu_split -> reuse it for scores + P
  float*  wScores  = (float*)wFusedRaw;                           // 67 MB
  bf16_t* wP       = (bf16_t*)(wFusedRaw + (size_t)BATCH * SEQ * SEQ * 4);

  cvt_f32_to_bf16<<<2048, 256, 0, stream>>>(expand, wExpand, (long)ETOT * DDIM);
  cvt_f32_to_bf16<<<2048, 256, 0, stream>>>(project, wProject, (long)DDIM * EXPDIM);

  layernorm_bf16<<<MROWS, 256, 0, stream>>>(x, gamma, beta, wH);

  // fused = h @ expand^T  : (8192 x 2048) x (8704 x 2048)^T
  gemm_wmma<0><<<dim3(ETOT / 128, MROWS / 128, 1), 256, 0, stream>>>(
      wH, wExpand, wFused, nullptr, MROWS, ETOT, DDIM, DDIM, DDIM, ETOT,
      0, 0, 0, 1.f, nullptr, nullptr, 0);

  geglu_split<<<MROWS, 256, 0, stream>>>(wFused, wQ, wK, wConcat, wValue);

  // scores = scale * Q @ K^T + causal ALiBi  (per batch)
  gemm_wmma<1><<<dim3(SEQ / 128, SEQ / 128, BATCH), 256, 0, stream>>>(
      wQ, wK, wScores, nullptr, SEQ, SEQ, QKDIM, QKDIM, QKDIM, SEQ,
      (long)SEQ * QKDIM, (long)SEQ * QKDIM, (long)SEQ * SEQ,
      0.0625f /* 1/sqrt(256) */, posm, nullptr, 0);

  softmax_rows<<<BATCH * SEQ, 256, 0, stream>>>(wScores, wP);

  // attn = P @ V -> concat[:, 2048:4096]  (per batch, causal-clamped K loop)
  gemm_wmma<2><<<dim3(DDIM / 128, SEQ / 128, BATCH), 256, 0, stream>>>(
      wP, wValue, nullptr, wConcat, SEQ, DDIM, SEQ, SEQ, DDIM, EXPDIM,
      (long)SEQ * SEQ, (long)SEQ * DDIM, (long)SEQ * EXPDIM,
      1.f, nullptr, nullptr, DDIM);

  // out = concat @ project^T + x
  gemm_wmma<3><<<dim3(DDIM / 128, MROWS / 128, 1), 256, 0, stream>>>(
      wConcat, wProject, out, nullptr, MROWS, DDIM, EXPDIM, EXPDIM, EXPDIM, DDIM,
      0, 0, 0, 1.f, nullptr, x, 0);
}
